// PianoMuseRWKV_42597485642147
// MI455X (gfx1250) — matrix operand order repaired and checked by reading, not compiled, not run on hardware
//
#include <hip/hip_runtime.h>
#include <hip/hip_bf16.h>

// RWKV-4 forward for gfx1250 (MI455X). GEMMs run on v_wmma_f32_16x16x32_bf16
// (bf16 A/B, f32 accum) with GLOBAL_LOAD_ASYNC_TO_LDS_B128 double-buffered
// staging (ASYNCcnt-tracked, no VGPR round-trip). ~580MB workspace.

typedef __attribute__((ext_vector_type(16))) __bf16 v16bf;
typedef __attribute__((ext_vector_type(8)))  __bf16 v8bf;
typedef __attribute__((ext_vector_type(8)))  float  v8f;

#define RW_L 8
#define RW_C 1024
#define RW_V 16384
#define RW_B 4
#define RW_T 2048
#define RW_F 4096
#define RW_M (RW_B * RW_T)   // 8192 tokens

__device__ __forceinline__ unsigned short f2bf(float f) {
  union { float f; unsigned u; } v; v.f = f;
  unsigned r = v.u + 0x7FFFu + ((v.u >> 16) & 1u);   // round-to-nearest-even
  return (unsigned short)(r >> 16);
}

__device__ __forceinline__ void wait_async0() {
#if __has_builtin(__builtin_amdgcn_s_wait_asynccnt)
  __builtin_amdgcn_s_wait_asynccnt(0);
#else
  asm volatile("s_wait_asynccnt 0" ::: "memory");
#endif
}

// async copy of 32 bytes (2 x b128) global -> LDS; offset applies to both sides
__device__ __forceinline__ void async_cp32(unsigned lds, const unsigned short* g) {
  asm volatile(
      "global_load_async_to_lds_b128 %0, %1, off\n\t"
      "global_load_async_to_lds_b128 %0, %1, off offset:16"
      :: "v"(lds), "v"(g) : "memory");
}
// async copy of 64 bytes (4 x b128)
__device__ __forceinline__ void async_cp64(unsigned lds, const unsigned short* g) {
  asm volatile(
      "global_load_async_to_lds_b128 %0, %1, off\n\t"
      "global_load_async_to_lds_b128 %0, %1, off offset:16\n\t"
      "global_load_async_to_lds_b128 %0, %1, off offset:32\n\t"
      "global_load_async_to_lds_b128 %0, %1, off offset:48"
      :: "v"(lds), "v"(g) : "memory");
}

__device__ __forceinline__ unsigned lds_addr(const void* p) {
  // generic pointer to LDS: low 32 bits are the LDS byte address (flat aperture)
  return (unsigned)(unsigned long long)p;
}

// ---------------------------------------------------------------- convert
__global__ void cvt_bf16_kernel(const float* __restrict__ in,
                                unsigned short* __restrict__ out, size_t n) {
  size_t i = (size_t)blockIdx.x * blockDim.x + threadIdx.x;
  size_t st = (size_t)gridDim.x * blockDim.x;
  for (; i < n; i += st) out[i] = f2bf(in[i]);
}

// ---------------------------------------------------------------- embed + ln0
__global__ __launch_bounds__(256) void embed_ln_kernel(
    const int* __restrict__ ids, const float* __restrict__ emb,
    const float* __restrict__ w, const float* __restrict__ b,
    float* __restrict__ x) {
  __shared__ float s0[256], s1[256];
  const int m = blockIdx.x;
  const int tid = threadIdx.x;
  const float* row = emb + (size_t)ids[m] * RW_C;
  float v[4]; float s = 0.f, q = 0.f;
#pragma unroll
  for (int e = 0; e < 4; ++e) { v[e] = row[tid * 4 + e]; s += v[e]; q += v[e] * v[e]; }
  s0[tid] = s; s1[tid] = q; __syncthreads();
  for (int st = 128; st > 0; st >>= 1) {
    if (tid < st) { s0[tid] += s0[tid + st]; s1[tid] += s1[tid + st]; }
    __syncthreads();
  }
  float mean = s0[0] * (1.f / RW_C);
  float var  = s1[0] * (1.f / RW_C) - mean * mean;
  float r = rsqrtf(var + 1e-5f);
  float* xo = x + (size_t)m * RW_C;
#pragma unroll
  for (int e = 0; e < 4; ++e) {
    int ch = tid * 4 + e;
    xo[ch] = (v[e] - mean) * r * w[ch] + b[ch];
  }
}

// ---------------------------------------------------- ln + token-shift + mix -> bf16
__global__ __launch_bounds__(256) void ln_mix_kernel(
    const float* __restrict__ x,
    const float* __restrict__ lnw, const float* __restrict__ lnb,
    const float* __restrict__ mk, const float* __restrict__ mv,
    const float* __restrict__ mr,
    unsigned short* __restrict__ ok, unsigned short* __restrict__ ov,
    unsigned short* __restrict__ orr) {
  __shared__ float s0[256], s1[256], s2[256], s3[256];
  const int m = blockIdx.x;
  const int t = m & (RW_T - 1);
  const int tid = threadIdx.x;
  const float* xc = x + (size_t)m * RW_C;
  const float* xp = xc - RW_C;   // only dereferenced when t > 0
  float c[4], p[4];
  float sc = 0, qc = 0, sp = 0, qp = 0;
#pragma unroll
  for (int e = 0; e < 4; ++e) {
    int ch = tid * 4 + e;
    c[e] = xc[ch]; sc += c[e]; qc += c[e] * c[e];
    float pv = (t == 0) ? 0.f : xp[ch];
    p[e] = pv; sp += pv; qp += pv * pv;
  }
  s0[tid] = sc; s1[tid] = qc; s2[tid] = sp; s3[tid] = qp; __syncthreads();
  for (int st = 128; st > 0; st >>= 1) {
    if (tid < st) {
      s0[tid] += s0[tid + st]; s1[tid] += s1[tid + st];
      s2[tid] += s2[tid + st]; s3[tid] += s3[tid + st];
    }
    __syncthreads();
  }
  float mc = s0[0] * (1.f / RW_C), rc = rsqrtf(s1[0] * (1.f / RW_C) - mc * mc + 1e-5f);
  float mp = s2[0] * (1.f / RW_C), rp = rsqrtf(s3[0] * (1.f / RW_C) - mp * mp + 1e-5f);
  size_t base = (size_t)m * RW_C;
#pragma unroll
  for (int e = 0; e < 4; ++e) {
    int ch = tid * 4 + e;
    float a  = (c[e] - mc) * rc * lnw[ch] + lnb[ch];
    float pn = (t == 0) ? 0.f : ((p[e] - mp) * rp * lnw[ch] + lnb[ch]);  // shift(LN(x))
    float fk = mk[ch]; ok[base + ch]  = f2bf(a * fk + pn * (1.f - fk));
    if (ov) { float fv = mv[ch]; ov[base + ch] = f2bf(a * fv + pn * (1.f - fv)); }
    float fr = mr[ch]; orr[base + ch] = f2bf(a * fr + pn * (1.f - fr));
  }
}

// ---------------------------------------------------------------- ln -> bf16
__global__ __launch_bounds__(256) void ln_to_bf16_kernel(
    const float* __restrict__ x, const float* __restrict__ w,
    const float* __restrict__ b, unsigned short* __restrict__ o) {
  __shared__ float s0[256], s1[256];
  const int m = blockIdx.x;
  const int tid = threadIdx.x;
  const float* xc = x + (size_t)m * RW_C;
  float v[4]; float s = 0.f, q = 0.f;
#pragma unroll
  for (int e = 0; e < 4; ++e) { v[e] = xc[tid * 4 + e]; s += v[e]; q += v[e] * v[e]; }
  s0[tid] = s; s1[tid] = q; __syncthreads();
  for (int st = 128; st > 0; st >>= 1) {
    if (tid < st) { s0[tid] += s0[tid + st]; s1[tid] += s1[tid + st]; }
    __syncthreads();
  }
  float mean = s0[0] * (1.f / RW_C);
  float r = rsqrtf(s1[0] * (1.f / RW_C) - mean * mean + 1e-5f);
  size_t base = (size_t)m * RW_C;
#pragma unroll
  for (int e = 0; e < 4; ++e) {
    int ch = tid * 4 + e;
    o[base + ch] = f2bf((v[e] - mean) * r * w[ch] + b[ch]);
  }
}

// ---------------------------------------------------------------- WMMA GEMM
// D[M,N] = A[M,K](bf16) x W[N,K](bf16)^T, f32 accumulate.
// Block: 256 threads = 8 waves, tile 128x256, BK=32, double-buffered LDS via
// GLOBAL_LOAD_ASYNC_TO_LDS_B128. Wave grid 2x4; wave tile 64x64 = 4x4 WMMAs.
#define EPI_NONE 0
#define EPI_SIG 1
#define EPI_RELUSQ 2

template <int EPI>
__global__ __launch_bounds__(256) void gemm_bf16_kernel(
    const unsigned short* __restrict__ A, const unsigned short* __restrict__ W,
    int M, int N, int K,
    const float* __restrict__ resid,
    float* __restrict__ outF, unsigned short* __restrict__ outB) {
  __shared__ unsigned short sA[2][128][32];   // 16KB
  __shared__ unsigned short sW[2][256][32];   // 32KB
  const int tid  = threadIdx.x;
  const int lane = tid & 31;
  const int wave = tid >> 5;
  const int wm = wave >> 2, wn = wave & 3;
  const int half = lane >> 4, l15 = lane & 15;
  const int bm = blockIdx.y * 128, bn = blockIdx.x * 256;

  v8f zero;
#pragma unroll
  for (int e = 0; e < 8; ++e) zero[e] = 0.f;
  v8f acc[4][4];
#pragma unroll
  for (int i = 0; i < 4; ++i)
#pragma unroll
    for (int j = 0; j < 4; ++j) acc[i][j] = zero;

  // async staging assignments:
  //  A tile 128x32 bf16 (8KB): thread t -> row t/2, 32B chunk (t%2)
  //  W tile 256x32 bf16 (16KB): thread t -> row t, full 64B row
  const int arow = tid >> 1;
  const int acol = (tid & 1) * 16;   // elements
  const unsigned short* gA = A + (size_t)(bm + arow) * K + acol;
  const unsigned short* gW = W + (size_t)(bn + tid) * K;
  unsigned la[2], lw[2];
  la[0] = lds_addr(&sA[0][arow][acol]); la[1] = lds_addr(&sA[1][arow][acol]);
  lw[0] = lds_addr(&sW[0][tid][0]);     lw[1] = lds_addr(&sW[1][tid][0]);

  auto stage = [&](int buf, int k0) {
    async_cp32(la[buf], gA + k0);
    async_cp64(lw[buf], gW + k0);
  };

  stage(0, 0);
  int cur = 0;
  for (int k0 = 0; k0 < K; k0 += 32) {
    wait_async0();        // our async writes for tile `cur` done
    __syncthreads();      // everyone's writes visible / previous reads done
    if (k0 + 32 < K) stage(cur ^ 1, k0 + 32);   // prefetch next tile into other buffer

    // A fragment: lanes 0-15 row M, K 0..15 ; lanes 16-31: K 16..31
    v16bf afrag[4], wfrag[4];
#pragma unroll
    for (int i = 0; i < 4; ++i) {
      int row = wm * 64 + i * 16 + l15;
      afrag[i] = *(const v16bf*)&sA[cur][row][half * 16];
    }
    // B fragment: lane = column N, 16 contiguous K values per half-wave
#pragma unroll
    for (int j = 0; j < 4; ++j) {
      int row = wn * 64 + j * 16 + l15;
      wfrag[j] = *(const v16bf*)&sW[cur][row][half * 16];
    }
#pragma unroll
    for (int i = 0; i < 4; ++i)
#pragma unroll
      for (int j = 0; j < 4; ++j)
        acc[i][j] = __builtin_amdgcn_wmma_f32_16x16x32_bf16(
            false, afrag[i], false, wfrag[j], (short)0, acc[i][j], false, false);
    cur ^= 1;
  }

  // epilogue: C/D layout -> lane l15 = column, vgpr r = row (+8 for upper half)
#pragma unroll
  for (int i = 0; i < 4; ++i) {
    int grow0 = bm + wm * 64 + i * 16 + half * 8;
#pragma unroll
    for (int j = 0; j < 4; ++j) {
      int gcol = bn + wn * 64 + j * 16 + l15;
#pragma unroll
      for (int r = 0; r < 8; ++r) {
        size_t idx = (size_t)(grow0 + r) * N + gcol;
        float s = acc[i][j][r];
        if (EPI == EPI_NONE) {
          if (resid) s += resid[idx];
          outF[idx] = s;
        } else if (EPI == EPI_SIG) {
          outF[idx] = 1.f / (1.f + __expf(-s));
        } else {  // relu^2 -> bf16 (feeds next GEMM)
          float tp = s > 0.f ? s : 0.f;
          outB[idx] = f2bf(tp * tp);
        }
      }
    }
  }
}

// ---------------------------------------------------------------- WKV scan
__global__ __launch_bounds__(256) void wkv_kernel(
    const float* __restrict__ k, const float* __restrict__ v,
    const float* __restrict__ td, const float* __restrict__ tf,
    float* __restrict__ out) {
  int gid = blockIdx.x * blockDim.x + threadIdx.x;   // 0 .. B*C-1
  int c = gid & (RW_C - 1);
  int b = gid >> 10;
  float w = -__expf(td[c]);
  float u = tf[c];
  float aa = 0.f, bb = 0.f, pp = -1e30f;
  const float* kp = k + (size_t)b * RW_T * RW_C + c;
  const float* vp = v + (size_t)b * RW_T * RW_C + c;
  float* op = out + (size_t)b * RW_T * RW_C + c;
  for (int t = 0; t < RW_T; ++t) {
    float kt = kp[(size_t)t * RW_C];
    float vt = vp[(size_t)t * RW_C];
    float ww = u + kt;
    float pmax = fmaxf(pp, ww);
    float e1 = __expf(pp - pmax);
    float e2 = __expf(ww - pmax);
    op[(size_t)t * RW_C] = (e1 * aa + e2 * vt) / (e1 * bb + e2);
    float ww2 = pp + w;
    float p2 = fmaxf(ww2, kt);
    float f1 = __expf(ww2 - p2);
    float f2 = __expf(kt - p2);
    aa = f1 * aa + f2 * vt;
    bb = f1 * bb + f2;
    pp = p2;
  }
}

// ---------------------------------------------------------------- elementwise
__global__ void mul_bf16_kernel(const float* __restrict__ a,
                                const float* __restrict__ b,
                                unsigned short* __restrict__ o, size_t n) {
  size_t i = (size_t)blockIdx.x * blockDim.x + threadIdx.x;
  size_t st = (size_t)gridDim.x * blockDim.x;
  for (; i < n; i += st) o[i] = f2bf(a[i] * b[i]);
}

__global__ void fma_add_kernel(float* __restrict__ x, const float* __restrict__ r,
                               const float* __restrict__ kv, size_t n) {
  size_t i = (size_t)blockIdx.x * blockDim.x + threadIdx.x;
  size_t st = (size_t)gridDim.x * blockDim.x;
  for (; i < n; i += st) x[i] += r[i] * kv[i];
}

// ---------------------------------------------------------------- launch
extern "C" void kernel_launch(void* const* d_in, const int* in_sizes, int n_in,
                              void* d_out, int out_size, void* d_ws, size_t ws_size,
                              hipStream_t stream) {
  (void)in_sizes; (void)n_in; (void)out_size; (void)ws_size;
  const int*   ids   = (const int*)d_in[0];
  const float* emb   = (const float*)d_in[1];
  const float* ln0w  = (const float*)d_in[2];
  const float* ln0b  = (const float*)d_in[3];
  const float* ln1w  = (const float*)d_in[4];
  const float* ln1b  = (const float*)d_in[5];
  const float* atmk  = (const float*)d_in[6];
  const float* atmv  = (const float*)d_in[7];
  const float* atmr  = (const float*)d_in[8];
  const float* atd   = (const float*)d_in[9];
  const float* atf   = (const float*)d_in[10];
  const float* awk   = (const float*)d_in[11];
  const float* awv   = (const float*)d_in[12];
  const float* awr   = (const float*)d_in[13];
  const float* awo   = (const float*)d_in[14];
  const float* ln2w  = (const float*)d_in[15];
  const float* ln2b  = (const float*)d_in[16];
  const float* ftmk  = (const float*)d_in[17];
  const float* ftmr  = (const float*)d_in[18];
  const float* fwk   = (const float*)d_in[19];
  const float* fwv   = (const float*)d_in[20];
  const float* fwr   = (const float*)d_in[21];
  const float* lnow  = (const float*)d_in[22];
  const float* lnob  = (const float*)d_in[23];
  const float* headw = (const float*)d_in[24];
  float* out = (float*)d_out;

  char* ws = (char*)d_ws;
  size_t off = 0;
  auto take = [&](size_t bytes) -> char* {
    char* p = ws + off;
    off += (bytes + 255) & ~(size_t)255;
    return p;
  };
  const size_t CC = (size_t)RW_C * RW_C;          // 1M
  const size_t FC = (size_t)RW_F * RW_C;          // 4M
  const size_t MC = (size_t)RW_M * RW_C;          // 8M
  const size_t MF = (size_t)RW_M * RW_F;          // 32M

  // bf16 weight copies
  unsigned short* bwk   = (unsigned short*)take(RW_L * CC * 2);
  unsigned short* bwv   = (unsigned short*)take(RW_L * CC * 2);
  unsigned short* bwr   = (unsigned short*)take(RW_L * CC * 2);
  unsigned short* bwo   = (unsigned short*)take(RW_L * CC * 2);
  unsigned short* bfk   = (unsigned short*)take(RW_L * FC * 2);
  unsigned short* bfv   = (unsigned short*)take(RW_L * FC * 2);
  unsigned short* bfr   = (unsigned short*)take(RW_L * CC * 2);
  unsigned short* bhead = (unsigned short*)take((size_t)RW_V * RW_C * 2);
  // activations
  float* x            = (float*)take(MC * 4);
  unsigned short* mk  = (unsigned short*)take(MC * 2);
  unsigned short* mv  = (unsigned short*)take(MC * 2);
  unsigned short* mr  = (unsigned short*)take(MC * 2);
  float* kb           = (float*)take(MC * 4);
  float* vb           = (float*)take(MC * 4);
  float* rb           = (float*)take(MC * 4);
  float* wkvb         = (float*)take(MC * 4);
  unsigned short* rwb = (unsigned short*)take(MC * 2);
  unsigned short* hb  = (unsigned short*)take(MF * 2);
  float* kvb          = (float*)take(MC * 4);
  unsigned short* xnb = (unsigned short*)take(MC * 2);

  auto cvt = [&](const float* s, unsigned short* d, size_t n) {
    cvt_bf16_kernel<<<1024, 256, 0, stream>>>(s, d, n);
  };
  cvt(awk, bwk, RW_L * CC);
  cvt(awv, bwv, RW_L * CC);
  cvt(awr, bwr, RW_L * CC);
  cvt(awo, bwo, RW_L * CC);
  cvt(fwk, bfk, RW_L * FC);
  cvt(fwv, bfv, RW_L * FC);
  cvt(fwr, bfr, RW_L * CC);
  cvt(headw, bhead, (size_t)RW_V * RW_C);

  embed_ln_kernel<<<RW_M, 256, 0, stream>>>(ids, emb, ln0w, ln0b, x);

  const dim3 blk(256);
  const dim3 gCC(RW_C / 256, RW_M / 128);
  const dim3 gFC(RW_F / 256, RW_M / 128);
  const dim3 gHV(RW_V / 256, RW_M / 128);

  for (int l = 0; l < RW_L; ++l) {
    size_t lc = (size_t)l * RW_C;
    // --- time mixing ---
    ln_mix_kernel<<<RW_M, blk, 0, stream>>>(x, ln1w + lc, ln1b + lc,
                                            atmk + lc, atmv + lc, atmr + lc,
                                            mk, mv, mr);
    gemm_bf16_kernel<EPI_NONE><<<gCC, blk, 0, stream>>>(
        mk, bwk + l * CC, RW_M, RW_C, RW_C, nullptr, kb, nullptr);
    gemm_bf16_kernel<EPI_NONE><<<gCC, blk, 0, stream>>>(
        mv, bwv + l * CC, RW_M, RW_C, RW_C, nullptr, vb, nullptr);
    gemm_bf16_kernel<EPI_SIG><<<gCC, blk, 0, stream>>>(
        mr, bwr + l * CC, RW_M, RW_C, RW_C, nullptr, rb, nullptr);
    wkv_kernel<<<(RW_B * RW_C) / 256, blk, 0, stream>>>(kb, vb, atd + lc, atf + lc, wkvb);
    mul_bf16_kernel<<<1024, blk, 0, stream>>>(rb, wkvb, rwb, MC);
    gemm_bf16_kernel<EPI_NONE><<<gCC, blk, 0, stream>>>(
        rwb, bwo + l * CC, RW_M, RW_C, RW_C, /*resid=*/x, x, nullptr);
    // --- channel mixing ---
    ln_mix_kernel<<<RW_M, blk, 0, stream>>>(x, ln2w + lc, ln2b + lc,
                                            ftmk + lc, nullptr, ftmr + lc,
                                            mk, nullptr, mr);
    gemm_bf16_kernel<EPI_RELUSQ><<<gFC, blk, 0, stream>>>(
        mk, bfk + (size_t)l * FC, RW_M, RW_F, RW_C, nullptr, nullptr, hb);
    gemm_bf16_kernel<EPI_NONE><<<gCC, blk, 0, stream>>>(
        hb, bfv + (size_t)l * FC, RW_M, RW_C, RW_F, nullptr, kvb, nullptr);
    gemm_bf16_kernel<EPI_SIG><<<gCC, blk, 0, stream>>>(
        mr, bfr + l * CC, RW_M, RW_C, RW_C, nullptr, rb, nullptr);
    fma_add_kernel<<<1024, blk, 0, stream>>>(x, rb, kvb, MC);
  }

  ln_to_bf16_kernel<<<RW_M, blk, 0, stream>>>(x, lnow, lnob, xnb);
  gemm_bf16_kernel<EPI_NONE><<<gHV, blk, 0, stream>>>(
      xnb, bhead, RW_M, RW_V, RW_C, nullptr, out, nullptr);
}
